// Vgg16c_72481868087446
// MI455X (gfx1250) — compile-verified, matrix operand order checked by reading, not run on hardware
//
#include <hip/hip_runtime.h>
#include <hip/hip_bf16.h>

// ---------------------------------------------------------------------------
// MI455X (gfx1250) implementation: all convs as implicit-GEMM on
// v_wmma_f32_16x16x32_bf16 (wave32, 16x16 tiles, fp32 accumulate).
// Block tile 64 Cout x 64 px, K-tile 64 -> 4 WMMA issues per LDS stage.
// ---------------------------------------------------------------------------

typedef __attribute__((ext_vector_type(16))) __bf16 v16bf;
typedef __attribute__((ext_vector_type(8)))  float  v8f;

#define ACT_NONE 0
#define ACT_RELU 1
#define NL_EPS 1e-4f

// ---------------------------------------------------------------------------
// Implicit-GEMM conv, batch 1, NCHW, stride 1, SAME pad, k=3x3 or 1x1.
//   A = Wt[Cout, K]  (K = Cin*9 or Cin), bf16 row-major
//   B = im2col(X)[K, H*W] gathered on the fly (H,W powers of two)
// Block: 256 thr = 8 wave32, wave grid 4(Cout) x 2(px); each wave computes
// 16 Cout x 32 px (two 16x16 WMMA tiles sharing the A fragment).
// K consumed in tiles of 64 (2 chunks of 32) staged in LDS directly in the
// 16-bit WMMA fragment layout: lane L holds row M=L&15, element e maps to
// k = (e&7) + (e>>3)*16 + (L>>4)*8.
// ---------------------------------------------------------------------------
__global__ __launch_bounds__(256) void conv_wmma(
    const __bf16* __restrict__ X,   // [Cin, H, W]
    const __bf16* __restrict__ Wt,  // [Cout, K]
    const float*  __restrict__ bias,// [Cout] or nullptr
    __bf16* __restrict__ Y,         // [Cout, H, W] bf16 (may be null)
    float*  __restrict__ Yf,        // optional fp32 output (overrides Y)
    int Cin, int Cout, int H, int W, int k3, int act)
{
    __shared__ __align__(32) __bf16 lA[2 * 2048];  // 2 chunks x 4 Cout tiles
    __shared__ __align__(32) __bf16 lB[2 * 2048];  // 2 chunks x 4 px tiles

    const int t    = threadIdx.x;
    const int lane = t & 31;
    const int wave = t >> 5;
    const int wrow = wave >> 1;   // 0..3  (Cout tile)
    const int wcol = wave & 1;    // 0..1  (pixel super-tile, 32 px)

    const int HWp = H * W;
    const int p0  = blockIdx.x * 64;
    const int co0 = blockIdx.y * 64;
    const int K   = Cin * (k3 ? 9 : 1);
    const int wsh = 31 - __builtin_clz((unsigned)W);   // W = 2^wsh
    const int wmask = W - 1;

    // ---- hoist loop-invariant fragment-layout decode into registers ------
    int  aKl[8]; long aRow[8];
    int  bKl[8], bY0[8], bX0[8], bPok[8];
#pragma unroll
    for (int r = 0; r < 8; ++r) {
        int idx  = t * 8 + r;
        int tile = idx >> 9;
        int ln   = (idx >> 4) & 31;
        int e    = idx & 15;
        int kl   = (e & 7) + ((e >> 3) << 4) + ((ln >> 4) << 3);
        // A: weights
        int m    = co0 + tile * 16 + (ln & 15);
        aKl[r]   = kl;
        aRow[r]  = (m < Cout) ? (long)m * K : -1;
        // B: im2col pixels
        int p    = p0 + tile * 16 + (ln & 15);
        bKl[r]   = kl;
        bPok[r]  = (p < HWp);
        bY0[r]   = p >> wsh;
        bX0[r]   = p & wmask;
    }

    v8f acc0 = {};
    v8f acc1 = {};

    for (int k0 = 0; k0 < K; k0 += 64) {
        if (k0 + 64 < K)
            __builtin_prefetch(Wt + (long)co0 * K + k0 + 64, 0, 0); // global_prefetch_b8

#pragma unroll
        for (int c = 0; c < 2; ++c) {
            const int kb = k0 + c * 32;
            // ---- stage A (weights): 8 bf16 per thread per chunk ---------
#pragma unroll
            for (int r = 0; r < 8; ++r) {
                int k = kb + aKl[r];
                __bf16 v = (__bf16)0.f;
                if (aRow[r] >= 0 && k < K) v = Wt[aRow[r] + k];
                lA[c * 2048 + t * 8 + r] = v;
            }
            // ---- stage B (im2col gather): 8 bf16 per thread per chunk ---
#pragma unroll
            for (int r = 0; r < 8; ++r) {
                int k = kb + bKl[r];
                __bf16 v = (__bf16)0.f;
                if (k < K && bPok[r]) {
                    int ci, dy, dx;
                    if (k3) { ci = k / 9; int r9 = k - ci * 9; int q = r9 / 3;
                              dy = q - 1; dx = r9 - q * 3 - 1; }
                    else    { ci = k; dy = 0; dx = 0; }
                    int y = bY0[r] + dy;
                    int x = bX0[r] + dx;
                    if ((unsigned)y < (unsigned)H && (unsigned)x < (unsigned)W)
                        v = X[(long)ci * HWp + (y << wsh) + x];
                }
                lB[c * 2048 + t * 8 + r] = v;
            }
        }
        __syncthreads();

        const v16bf* Af = (const v16bf*)lA;
        const v16bf* Bf = (const v16bf*)lB;
#pragma unroll
        for (int c = 0; c < 2; ++c) {
            v16bf a  = Af[(c * 4 + wrow) * 32 + lane];          // shared A frag
            v16bf b0 = Bf[(c * 4 + wcol * 2 + 0) * 32 + lane];
            v16bf b1 = Bf[(c * 4 + wcol * 2 + 1) * 32 + lane];
            acc0 = __builtin_amdgcn_wmma_f32_16x16x32_bf16(
                false, a, false, b0, (short)0, acc0, false, false);
            acc1 = __builtin_amdgcn_wmma_f32_16x16x32_bf16(
                false, a, false, b1, (short)0, acc1, false, false);
        }
        __syncthreads();
    }

    // D layout: lanes 0-15 -> M = v, lanes 16-31 -> M = v+8; N = lane&15
    const int n     = lane & 15;
    const int mbase = co0 + wrow * 16 + ((lane < 16) ? 0 : 8);
#pragma unroll
    for (int j = 0; j < 2; ++j) {
        const int p = p0 + (wcol * 2 + j) * 16 + n;
        if (p < HWp) {
            const v8f acc = j ? acc1 : acc0;
#pragma unroll
            for (int v = 0; v < 8; ++v) {
                int m = mbase + v;
                if (m < Cout) {
                    float r = acc[v];
                    if (bias) r += bias[m];
                    if (act == ACT_RELU) r = fmaxf(r, 0.f);
                    if (Yf) Yf[(long)m * HWp + p] = r;
                    else    Y [(long)m * HWp + p] = (__bf16)r;
                }
            }
        }
    }
}

// ---------------------------------------------------------------------------
// Elementwise / data-movement kernels (bandwidth-trivial on 23.3 TB/s HBM)
// ---------------------------------------------------------------------------
__global__ void k_cvt_f32_bf16(const float* __restrict__ in, __bf16* __restrict__ out, long n) {
    long i = (long)blockIdx.x * blockDim.x + threadIdx.x;
    if (i < n) out[i] = (__bf16)in[i];
}

__global__ void k_add_bf16(const __bf16* __restrict__ a, const __bf16* __restrict__ b,
                           __bf16* __restrict__ o, long n) {
    long i = (long)blockIdx.x * blockDim.x + threadIdx.x;
    if (i < n) o[i] = (__bf16)((float)a[i] + (float)b[i]);
}

__global__ void k_maxpool2(const __bf16* __restrict__ in, __bf16* __restrict__ out,
                           int C, int H, int W) {
    int Ho = H >> 1, Wo = W >> 1;
    long n = (long)C * Ho * Wo;
    long i = (long)blockIdx.x * blockDim.x + threadIdx.x;
    if (i >= n) return;
    int c = (int)(i / (Ho * Wo));
    int r = (int)(i % (Ho * Wo));
    int h = r / Wo, w = r % Wo;
    const __bf16* p = in + ((long)c * H + 2 * h) * W + 2 * w;
    float m = (float)p[0];
    m = fmaxf(m, (float)p[1]);
    m = fmaxf(m, (float)p[W]);
    m = fmaxf(m, (float)p[W + 1]);
    out[i] = (__bf16)m;
}

// pixel_shuffle r=2: in [4C,H,W] -> out [C,2H,2W]; in ch = c*4 + (oh&1)*2 + (ow&1)
__global__ void k_pixshuf2(const __bf16* __restrict__ in, __bf16* __restrict__ out,
                           int C, int H, int W) {
    long n = (long)C * 4 * H * W;
    long i = (long)blockIdx.x * blockDim.x + threadIdx.x;
    if (i >= n) return;
    int HW4 = 4 * H * W;
    int c  = (int)(i / HW4);
    int r  = (int)(i % HW4);
    int oh = r / (2 * W), ow = r % (2 * W);
    int ic = c * 4 + (oh & 1) * 2 + (ow & 1);
    out[i] = in[((long)ic * H + (oh >> 1)) * W + (ow >> 1)];
}

// NLCN normalize + dynamic-kernel apply.
// w: [Cout, 9*Cin, H, W] predictor output (pre-activation).
// mode 0: a = relu(w)+eps, norm by sum(a);  mode 1: a = sign(w)(|w|+eps), norm by sum|a|.
// xc(k): block b=k/Cin gives roll (j=b%3-1 on H, i=b/3-1 on W); ci=k%Cin.
__global__ void k_nl_apply(const __bf16* __restrict__ x, const __bf16* __restrict__ w,
                           __bf16* __restrict__ y, int Cin, int Cout, int H, int W,
                           int signed_mode) {
    int HWp = H * W;
    long n = (long)Cout * HWp;
    long i = (long)blockIdx.x * blockDim.x + threadIdx.x;
    if (i >= n) return;
    int o = (int)(i / HWp);
    int p = (int)(i % HWp);
    int h = p / W, wq = p % W;
    int por = 9 * Cin;
    const __bf16* wp = w + (long)o * por * HWp + p;

    float s = 0.f;
    for (int k = 0; k < por; ++k) {
        float v = (float)wp[(long)k * HWp];
        float a = signed_mode ? copysignf(fabsf(v) + NL_EPS, v)
                              : fmaxf(v, 0.f) + NL_EPS;
        s += signed_mode ? fabsf(a) : a;
    }
    float inv = 1.f / s;
    float accv = 0.f;
    for (int k = 0; k < por; ++k) {
        float v = (float)wp[(long)k * HWp];
        float a = signed_mode ? copysignf(fabsf(v) + NL_EPS, v)
                              : fmaxf(v, 0.f) + NL_EPS;
        int b  = k / Cin, ci = k % Cin;
        int jj = b % 3 - 1;            // roll on H
        int ii = b / 3 - 1;            // roll on W
        int sh = h - jj;  sh += (sh < 0) ? H : 0;  sh -= (sh >= H) ? H : 0;
        int sw = wq - ii; sw += (sw < 0) ? W : 0;  sw -= (sw >= W) ? W : 0;
        accv += (a * inv) * (float)x[(long)ci * HWp + sh * W + sw];
    }
    y[i] = (__bf16)accv;
}

// ---------------------------------------------------------------------------
// Host orchestration
// ---------------------------------------------------------------------------
static inline long cdiv(long a, long b) { return (a + b - 1) / b; }

extern "C" void kernel_launch(void* const* d_in, const int* in_sizes, int n_in,
                              void* d_out, int out_size, void* d_ws, size_t ws_size,
                              hipStream_t stream) {
    (void)out_size; (void)ws_size;
    hipStream_t s = stream;
    char* ws = (char*)d_ws;
    size_t off = 0;
    auto alloc = [&](long elems, size_t elsz) -> char* {
        char* p = ws + off;
        off += ((size_t)elems * elsz + 255) & ~(size_t)255;
        return p;
    };

    // ---- convert all GEMM weights (fp32 -> bf16) ------------------------
    // d_in layout: 0=x; 1..26 = v{i}_w, v{i}_b; 27..32 = f1a,f1b,f1c,f3a,f3b,f3c;
    // 33..50 = {sk1,sk2,cl2,skcl2,cl3,skcl3,cl4,skcl4,cl5} (w,b) pairs.
    __bf16* bw[51] = {};
    for (int i = 1; i < n_in && i < 51; ++i) {
        bool isw = (i >= 1 && i <= 26 && ((i - 1) & 1) == 0) ||
                   (i >= 27 && i <= 32) ||
                   (i >= 33 && i <= 50 && ((i - 33) & 1) == 0);
        if (!isw) continue;
        long n = in_sizes[i];
        bw[i] = (__bf16*)alloc(n, 2);
        k_cvt_f32_bf16<<<(unsigned)cdiv(n, 256), 256, 0, s>>>((const float*)d_in[i], bw[i], n);
    }
    // input image -> bf16
    __bf16* xb = (__bf16*)alloc(3L * 128 * 128, 2);
    k_cvt_f32_bf16<<<(unsigned)cdiv(3L * 128 * 128, 256), 256, 0, s>>>(
        (const float*)d_in[0], xb, 3L * 128 * 128);

    auto convL = [&](const __bf16* X, const __bf16* Wt, const float* bias,
                     __bf16* Y, float* Yf, int Cin, int Cout, int H, int W,
                     int k3, int act) {
        dim3 g((unsigned)cdiv((long)H * W, 64), (unsigned)cdiv(Cout, 64));
        conv_wmma<<<g, 256, 0, s>>>(X, Wt, bias, Y, Yf, Cin, Cout, H, W, k3, act);
    };

    // ---- VGG encoder ----------------------------------------------------
    __bf16* pA  = (__bf16*)alloc(64L * 128 * 128, 2);
    __bf16* pB  = (__bf16*)alloc(64L * 128 * 128, 2);
    __bf16* xc4 = (__bf16*)alloc(64L  * 128 * 128, 2);
    __bf16* xc3 = (__bf16*)alloc(128L * 64 * 64, 2);
    __bf16* xc2 = (__bf16*)alloc(256L * 32 * 32, 2);
    __bf16* xc1 = (__bf16*)alloc(512L * 16 * 16, 2);

    const int cfg_ci[13] = {3, 64, 64, 128, 128, 256, 256, 256, 512, 512, 512, 512, 512};
    const int cfg_co[13] = {64, 64, 128, 128, 256, 256, 256, 512, 512, 512, 512, 512, 512};

    __bf16* cur = xb;
    int Hh = 128;
    for (int i = 0; i < 13; ++i) {
        __bf16* outp;
        if      (i == 1) outp = xc4;
        else if (i == 3) outp = xc3;
        else if (i == 6) outp = xc2;
        else if (i == 9) outp = xc1;
        else             outp = (cur == pA) ? pB : pA;
        convL(cur, bw[1 + 2 * i], (const float*)d_in[2 + 2 * i], outp, nullptr,
              cfg_ci[i], cfg_co[i], Hh, Hh, /*k3=*/1, ACT_RELU);
        cur = outp;
        if (i == 1 || i == 3 || i == 6 || i == 9) {
            __bf16* pp = (cur == pA) ? pB : pA;
            long n = (long)cfg_co[i] * (Hh / 2) * (Hh / 2);
            k_maxpool2<<<(unsigned)cdiv(n, 256), 256, 0, s>>>(cur, pp, cfg_co[i], Hh, Hh);
            Hh >>= 1;
            cur = pp;
        }
    }
    // cur = feats: 512 ch @ 8x8

    // ---- decoder scratch ------------------------------------------------
    __bf16* wscr  = (__bf16*)alloc(2880L * 64 * 64, 2);  // NLCN predictor output
    __bf16* cbuf1 = (__bf16*)alloc(20L * 64 * 64, 2);    // concat path 1
    __bf16* cbuf2 = (__bf16*)alloc(20L * 64 * 64, 2);    // concat path 2
    __bf16* psin  = (__bf16*)alloc(32L * 64 * 64, 2);    // pixel-shuffle input
    __bf16* ybuf  = (__bf16*)alloc(8L * 128 * 128, 2);
    __bf16* ybuf2 = (__bf16*)alloc(8L * 128 * 128, 2);
    __bf16* dec1  = (__bf16*)alloc(64L * 128 * 128, 2);
    __bf16* dec2  = (__bf16*)alloc(64L * 128 * 128, 2);

    auto nlapply = [&](const __bf16* x, const __bf16* w, __bf16* y,
                       int Cin, int Cout, int H, int mode) {
        long n = (long)Cout * H * H;
        k_nl_apply<<<(unsigned)cdiv(n, 256), 256, 0, s>>>(x, w, y, Cin, Cout, H, H, mode);
    };

    // One NLCN block: x (8ch,HxH) -> out (8ch, 2Hx2H)
    auto nlcn = [&](const __bf16* xin, int H, __bf16* out) {
        int HW = H * H;
        const __bf16 *f1a = bw[27], *f1b = bw[28], *f1c = bw[29];
        const __bf16 *f3a = bw[30], *f3b = bw[31], *f3c = bw[32];
        // path 1 (relu-normalized)
        hipMemcpyAsync(cbuf1, xin, (size_t)8 * HW * 2, hipMemcpyDeviceToDevice, s);
        convL(xin,   f1a, nullptr, wscr, nullptr, 8,  576,  H, H, 1, ACT_NONE);
        nlapply(xin,   wscr, cbuf1 + 8L * HW,  8, 8, H, 0);
        convL(cbuf1, f1b, nullptr, wscr, nullptr, 16, 576,  H, H, 1, ACT_NONE);
        nlapply(cbuf1, wscr, cbuf1 + 16L * HW, 16, 4, H, 0);
        convL(cbuf1, f1c, nullptr, wscr, nullptr, 20, 2880, H, H, 1, ACT_NONE);
        nlapply(cbuf1, wscr, psin, 20, 16, H, 0);
        // path 2 (signed-normalized)
        hipMemcpyAsync(cbuf2, xin, (size_t)8 * HW * 2, hipMemcpyDeviceToDevice, s);
        convL(xin,   f3a, nullptr, wscr, nullptr, 8,  576,  H, H, 1, ACT_NONE);
        nlapply(xin,   wscr, cbuf2 + 8L * HW,  8, 8, H, 1);
        convL(cbuf2, f3b, nullptr, wscr, nullptr, 16, 576,  H, H, 1, ACT_NONE);
        nlapply(cbuf2, wscr, cbuf2 + 16L * HW, 16, 4, H, 1);
        convL(cbuf2, f3c, nullptr, wscr, nullptr, 20, 2880, H, H, 1, ACT_NONE);
        nlapply(cbuf2, wscr, psin + 16L * HW, 20, 16, H, 1);
        // pixel shuffle 32ch,HxH -> 8ch,2Hx2H
        long n = 32L * HW;
        k_pixshuf2<<<(unsigned)cdiv(n, 256), 256, 0, s>>>(psin, out, 8, H, H);
    };

    auto addL = [&](const __bf16* a, const __bf16* b, __bf16* o, long n) {
        k_add_bf16<<<(unsigned)cdiv(n, 256), 256, 0, s>>>(a, b, o, n);
    };

    // ---- decoder --------------------------------------------------------
    // sk1: 512->8 @8
    convL(cur, bw[33], (const float*)d_in[34], ybuf, nullptr, 512, 8, 8, 8, 0, ACT_NONE);
    nlcn(ybuf, 8, ybuf2);                                               // -> 8ch @16
    convL(ybuf2, bw[35], (const float*)d_in[36], dec1, nullptr, 8, 512, 16, 16, 0, ACT_NONE);
    addL(dec1, xc1, dec2, 512L * 16 * 16);
    convL(dec2, bw[37], (const float*)d_in[38], ybuf, nullptr, 512, 8, 16, 16, 0, ACT_NONE);
    nlcn(ybuf, 16, ybuf2);                                              // -> 8ch @32
    convL(ybuf2, bw[39], (const float*)d_in[40], dec1, nullptr, 8, 256, 32, 32, 0, ACT_NONE);
    addL(dec1, xc2, dec2, 256L * 32 * 32);
    convL(dec2, bw[41], (const float*)d_in[42], ybuf, nullptr, 256, 8, 32, 32, 0, ACT_NONE);
    nlcn(ybuf, 32, ybuf2);                                              // -> 8ch @64
    convL(ybuf2, bw[43], (const float*)d_in[44], dec1, nullptr, 8, 128, 64, 64, 0, ACT_NONE);
    addL(dec1, xc3, dec2, 128L * 64 * 64);
    convL(dec2, bw[45], (const float*)d_in[46], ybuf, nullptr, 128, 8, 64, 64, 0, ACT_NONE);
    nlcn(ybuf, 64, ybuf2);                                              // -> 8ch @128
    convL(ybuf2, bw[47], (const float*)d_in[48], dec1, nullptr, 8, 64, 128, 128, 0, ACT_NONE);
    addL(dec1, xc4, dec2, 64L * 128 * 128);
    // cl5: 64->1, fp32 directly to d_out
    convL(dec2, bw[49], (const float*)d_in[50], nullptr, (float*)d_out, 64, 1, 128, 128, 0, ACT_NONE);
}
